// AstroSymbolicEpisodicLayer_85461259256335
// MI455X (gfx1250) — matrix operand order repaired and checked by reading, not compiled, not run on hardware
//
#include <hip/hip_runtime.h>
#include <math.h>
#include <stdint.h>

typedef __bf16 bf16;
typedef __bf16 v16bf __attribute__((ext_vector_type(16)));
typedef __bf16 v8bf  __attribute__((ext_vector_type(8)));
typedef __bf16 v4bf  __attribute__((ext_vector_type(4)));
typedef float  v8f   __attribute__((ext_vector_type(8)));

#define B_SZ 4
#define T_Q 2048
#define T_K 2048
#define D_M 1024
#define R_R 64
#define H_H 512
#define N_TOK (B_SZ * T_Q)

enum { EPI_NONE = 0, EPI_BIAS, EPI_BIAS_RELU, EPI_SCORE, EPI_ROWSCALE };

// ---------------------------------------------------------------------------
// CDNA5 async global->LDS copy (ASYNCcnt path). Per CDNA5 ISA:
//   VDST VGPR = per-lane LDS byte address, VADDR = 64-bit global address.
// Generic shared pointers carry the LDS byte offset in their low 32 bits
// (flat->LDS aperture truncation), so (u32)(size_t)p is the LDS address.
// ---------------------------------------------------------------------------
__device__ __forceinline__ void async_cp16(void* lds, const void* gaddr) {
  asm volatile("global_load_async_to_lds_b128 %0, %1, off"
               :: "v"((uint32_t)(size_t)lds),
                  "v"((unsigned long long)(size_t)gaddr)
               : "memory");
}
__device__ __forceinline__ void async_wait0() {
  asm volatile("s_wait_asynccnt 0x0" ::: "memory");
}

// ---------------------------------------------------------------------------
// Generic WMMA bf16 GEMM: C = epi(A @ B [+ bias])
//   A: row-major M x K, f32 or bf16 (A_BF16)
//   B: bf16; B_IS_T ? stored (N x K) : stored (K x N)
//   256 threads = 8 waves (4x2), WG tile 128x64, wave tile 32x32, K-step 64.
// All dims must be multiples of (128, 64, 64) — true for every call below.
// ---------------------------------------------------------------------------
template<bool A_BF16, bool B_IS_T, bool OUT_BF16, int EPI>
__global__ __launch_bounds__(256) void gemm_wmma(
    const void* __restrict__ Ap, const bf16* __restrict__ Bp, void* __restrict__ Cp,
    int M, int N, int K,
    long sAb, long sBb, long sCb,                 // per-batch element strides
    const float* __restrict__ bias, const float* __restrict__ tau,
    const float* __restrict__ rscale, int rsStride)
{
  __shared__ alignas(16) bf16 sA[128 * 72];
  __shared__ alignas(16) bf16 sB[64 * 72];
  const int tid  = threadIdx.x;
  const int lane = tid & 31, wid = tid >> 5;
  const int wm = wid >> 1, wn = wid & 1;
  const int nb = blockIdx.x, mb = blockIdx.y, bz = blockIdx.z;
  const long aBase = (long)bz * sAb + (long)mb * 128 * K;
  const long bBase = (long)bz * sBb;

  v8f acc[2][2] = {};

  for (int k0 = 0; k0 < K; k0 += 64) {
    // ---- stage A tile 128x64 into LDS ----
    if (A_BF16) {
      // pure bf16 copy: async DMA, 16B per lane-op, 4 ops/thread
      const bf16* Ab = (const bf16*)Ap;
      #pragma unroll
      for (int u = 0; u < 4; u++) {
        int i = tid * 8 + u * 2048;
        int m = i >> 6, kk = i & 63;
        async_cp16(&sA[m * 72 + kk], &Ab[aBase + (long)m * K + (k0 + kk)]);
      }
    } else {
      // f32 -> bf16 conversion: issue ALL 8 b128 loads first (overlap HBM
      // latency), then convert + packed b64 LDS stores.
      const float* Af = (const float*)Ap;
      float4 t[8];
      #pragma unroll
      for (int u = 0; u < 8; u++) {
        int i = tid * 4 + u * 1024;
        int m = i >> 6, kk = i & 63;
        t[u] = *(const float4*)&Af[aBase + (long)m * K + (k0 + kk)];
      }
      #pragma unroll
      for (int u = 0; u < 8; u++) {
        int i = tid * 4 + u * 1024;
        int m = i >> 6, kk = i & 63;
        v4bf o = { (bf16)t[u].x, (bf16)t[u].y, (bf16)t[u].z, (bf16)t[u].w };
        *(v4bf*)&sA[m * 72 + kk] = o;
      }
    }
    // ---- stage B tile into LDS as [n][k] ----
    if (B_IS_T) {
      // already (N x K): straight row copy via async DMA, 2 ops/thread
      #pragma unroll
      for (int u = 0; u < 2; u++) {
        int i = tid * 8 + u * 2048;
        int nn = i >> 6, kk = i & 63;
        async_cp16(&sB[nn * 72 + kk], &Bp[bBase + (long)(nb * 64 + nn) * K + (k0 + kk)]);
      }
    } else {
      // (K x N): two b128 global reads along N in flight, then LDS transpose
      v8bf t0, t1;
      {
        int i0 = tid * 8,        k0r = i0 >> 6, n0 = i0 & 63;
        int i1 = tid * 8 + 2048, k1r = i1 >> 6, n1 = i1 & 63;
        t0 = *(const v8bf*)&Bp[bBase + (long)(k0 + k0r) * N + (nb * 64 + n0)];
        t1 = *(const v8bf*)&Bp[bBase + (long)(k0 + k1r) * N + (nb * 64 + n1)];
        #pragma unroll
        for (int u = 0; u < 8; u++) sB[(n0 + u) * 72 + k0r] = t0[u];
        #pragma unroll
        for (int u = 0; u < 8; u++) sB[(n1 + u) * 72 + k1r] = t1[u];
      }
    }
    // prefetch next A tile (global_prefetch_b8)
    if (k0 + 64 < K) {
      long pidx = aBase + (long)(tid & 127) * K + (k0 + 64);
      const char* pa = A_BF16 ? (const char*)&((const bf16*)Ap)[pidx]
                              : (const char*)&((const float*)Ap)[pidx];
      __builtin_prefetch(pa, 0, 1);
    }
    if (A_BF16 || B_IS_T) async_wait0();   // own async copies complete
    __syncthreads();                        // cross-wave visibility

    const int hi = lane >> 4, lr = lane & 15;
    #pragma unroll
    for (int kk = 0; kk < 64; kk += 32) {
      v16bf af[2], bfm[2];
      #pragma unroll
      for (int t = 0; t < 2; t++) {
        // A 16x32 layout: halves 0..7 -> K = kk+hi*8.., halves 8..15 -> K = kk+16+hi*8..
        union { v16bf v; v8bf h[2]; } ua, ub;
        const bf16* ap = &sA[(wm * 32 + t * 16 + lr) * 72 + kk + hi * 8];
        ua.h[0] = *(const v8bf*)ap;
        ua.h[1] = *(const v8bf*)(ap + 16);
        af[t] = ua.v;
        // B 32x16 layout: lane n = lr, halves 0..7 -> K = kk+hi*8.., halves 8..15 -> K = kk+16+hi*8..
        const bf16* bp = &sB[(wn * 32 + t * 16 + lr) * 72 + kk + hi * 8];
        ub.h[0] = *(const v8bf*)bp;
        ub.h[1] = *(const v8bf*)(bp + 16);
        bfm[t] = ub.v;
      }
      #pragma unroll
      for (int i2 = 0; i2 < 2; i2++)
        #pragma unroll
        for (int j2 = 0; j2 < 2; j2++)
          acc[i2][j2] = __builtin_amdgcn_wmma_f32_16x16x32_bf16(
              false, af[i2], false, bfm[j2], (short)0, acc[i2][j2], false, false);
    }
    __syncthreads();
  }

  // ---- epilogue ----
  const float tauv = (EPI == EPI_SCORE) ? tau[bz] : 0.f;
  #pragma unroll
  for (int i2 = 0; i2 < 2; i2++)
  #pragma unroll
  for (int j2 = 0; j2 < 2; j2++) {
    int n = nb * 64 + wn * 32 + j2 * 16 + (lane & 15);
    float bv = (EPI == EPI_BIAS || EPI == EPI_BIAS_RELU) ? bias[n] : 0.f;
    #pragma unroll
    for (int r = 0; r < 8; r++) {
      int m = mb * 128 + wm * 32 + i2 * 16 + (lane >> 4) * 8 + r;
      float v = acc[i2][j2][r] + bv;
      if (EPI == EPI_BIAS_RELU) v = fmaxf(v, 0.f);
      if (EPI == EPI_SCORE) {            // Epanechnikov: clip(1 - tau*((1-cos)/2)^2, 0)
        float om = (1.f - v) * 0.5f;
        v = fmaxf(1.f - tauv * om * om, 0.f);
      }
      if (EPI == EPI_ROWSCALE) v *= rscale[bz * rsStride + m];
      long ci = (long)bz * sCb + (long)m * N + n;
      if (OUT_BF16) ((bf16*)Cp)[ci] = (bf16)v;
      else          ((float*)Cp)[ci] = v;
    }
  }
}

// ---------------------------------------------------------------------------
// Small helper kernels
// ---------------------------------------------------------------------------
__global__ void k_convert(const float* __restrict__ in, bf16* __restrict__ out, long n) {
  long i = ((long)blockIdx.x * blockDim.x + threadIdx.x) * 4;
  if (i < n) {
    float4 v = *(const float4*)&in[i];
    v4bf o = { (bf16)v.x, (bf16)v.y, (bf16)v.z, (bf16)v.w };
    *(v4bf*)&out[i] = o;
  }
}

__global__ void k_zero(float* p, int n) {
  int i = blockIdx.x * blockDim.x + threadIdx.x;
  if (i < n) p[i] = 0.f;
}

// DFT matrices: fwd (1024 x 2048) = [C | -S],  inv (2048 x 1024) = [C ; -S] / 1024
__global__ __launch_bounds__(256) void k_gen_dft(bf16* __restrict__ fwd, bf16* __restrict__ inv) {
  const float w = 6.28318530717958647692f / 1024.f;
  int b = blockIdx.x;
  if (b < 1024) {
    int m = b;
    for (int j = threadIdx.x; j < 2048; j += 256) {
      int jj = j & 1023;
      int p = (m * jj) & 1023;                    // exact phase mod N
      float ang = w * (float)p;
      float v = (j < 1024) ? cosf(ang) : -sinf(ang);
      fwd[(long)m * 2048 + j] = (bf16)v;
    }
  } else {
    int k = b - 1024;                              // 0..2047
    int kk = k & 1023;
    for (int n = threadIdx.x; n < 1024; n += 256) {
      int p = (kk * n) & 1023;
      float ang = w * (float)p;
      float v = (k < 1024) ? cosf(ang) : -sinf(ang);
      inv[(long)k * 1024 + n] = (bf16)(v * (1.f / 1024.f));
    }
  }
}

__global__ __launch_bounds__(256) void k_role_norm(const float* __restrict__ role, bf16* __restrict__ out) {
  __shared__ float red[256];
  int r = blockIdx.x;
  float s = 0.f;
  for (int d = threadIdx.x; d < D_M; d += 256) { float v = role[r * D_M + d]; s += v * v; }
  red[threadIdx.x] = s; __syncthreads();
  for (int o = 128; o > 0; o >>= 1) { if (threadIdx.x < o) red[threadIdx.x] += red[threadIdx.x + o]; __syncthreads(); }
  float inv = 1.f / fmaxf(sqrtf(red[0]), 1e-12f);
  for (int d = threadIdx.x; d < D_M; d += 256) out[r * D_M + d] = (bf16)(role[r * D_M + d] * inv);
}

// l2-normalize each 1024-row; optionally accumulate surprise sum per batch
__global__ __launch_bounds__(256) void k_rownorm(const float* __restrict__ in, bf16* __restrict__ out,
                                                 float* ssum, int rowsPerBatch) {
  __shared__ float red[256];
  long row = blockIdx.x;
  const float* p = in + row * D_M;
  float4 v0 = *(const float4*)&p[threadIdx.x * 4];
  float s = v0.x * v0.x + v0.y * v0.y + v0.z * v0.z + v0.w * v0.w;
  red[threadIdx.x] = s; __syncthreads();
  for (int o = 128; o > 0; o >>= 1) { if (threadIdx.x < o) red[threadIdx.x] += red[threadIdx.x + o]; __syncthreads(); }
  float nrm = sqrtf(red[0]);
  float inv = 1.f / fmaxf(nrm, 1e-12f);
  v4bf o = { (bf16)(v0.x * inv), (bf16)(v0.y * inv), (bf16)(v0.z * inv), (bf16)(v0.w * inv) };
  *(v4bf*)&out[row * D_M + threadIdx.x * 4] = o;
  if (ssum != nullptr && threadIdx.x == 0)
    atomicAdd(&ssum[row / rowsPerBatch], nrm * (1.f / 32.f));   // /sqrt(D)
}

__global__ void k_state(const float* __restrict__ ssum, const float* __restrict__ astro,
                        const float* __restrict__ ascale, float* __restrict__ new_state,
                        float* __restrict__ tau) {
  int b = threadIdx.x;
  if (b < B_SZ) {
    float mean = ssum[b] * (1.f / (float)T_Q);
    float ns = 0.95f * astro[b] + 0.05f * mean;
    new_state[b] = ns;
    tau[b] = 1.0f * fmaxf(1.f + ascale[0] * ns, 0.001f);
  }
}

__global__ __launch_bounds__(64) void k_softmax64(const float* __restrict__ logits, bf16* __restrict__ w) {
  __shared__ float red[64];
  long row = blockIdx.x;
  float v = logits[row * R_R + threadIdx.x];
  red[threadIdx.x] = v; __syncthreads();
  for (int o = 32; o > 0; o >>= 1) { if (threadIdx.x < o) red[threadIdx.x] = fmaxf(red[threadIdx.x], red[threadIdx.x + o]); __syncthreads(); }
  float mx = red[0]; __syncthreads();
  float e = expf(v - mx);
  red[threadIdx.x] = e; __syncthreads();
  for (int o = 32; o > 0; o >>= 1) { if (threadIdx.x < o) red[threadIdx.x] += red[threadIdx.x + o]; __syncthreads(); }
  w[row * R_R + threadIdx.x] = (bf16)(e / red[0]);
}

// P = KF * RF (complex, per row: cols 0..1023 = re, 1024..2047 = im)
__global__ __launch_bounds__(256) void k_cmul(const bf16* __restrict__ KF, const bf16* __restrict__ RF,
                                              bf16* __restrict__ P) {
  long i = (long)blockIdx.x * 256 + threadIdx.x;   // N_TOK * 1024 threads
  long row = i >> 10; int j = (int)(i & 1023);
  long base = row * 2048;
  float kre = (float)KF[base + j], kim = (float)KF[base + 1024 + j];
  float rre = (float)RF[base + j], rim = (float)RF[base + 1024 + j];
  P[base + j]        = (bf16)(kre * rre - kim * rim);
  P[base + 1024 + j] = (bf16)(kre * rim + kim * rre);
}

__global__ __launch_bounds__(256) void k_rowscale(const bf16* __restrict__ sc, float* __restrict__ rs) {
  __shared__ float red[256];
  long row = blockIdx.x;
  const bf16* p = sc + row * T_K;
  v8bf v = *(const v8bf*)&p[threadIdx.x * 8];
  float s = 0.f;
  #pragma unroll
  for (int u = 0; u < 8; u++) s += (float)v[u];
  red[threadIdx.x] = s; __syncthreads();
  for (int o = 128; o > 0; o >>= 1) { if (threadIdx.x < o) red[threadIdx.x] += red[threadIdx.x + o]; __syncthreads(); }
  if (threadIdx.x == 0) rs[row] = 1.f / fmaxf(red[0], 1e-9f);
}

// ---------------------------------------------------------------------------
extern "C" void kernel_launch(void* const* d_in, const int* in_sizes, int n_in,
                              void* d_out, int out_size, void* d_ws, size_t ws_size,
                              hipStream_t stream) {
  const float* q_in  = (const float*)d_in[0];
  const float* k_in  = (const float*)d_in[1];
  const float* v_in  = (const float*)d_in[2];
  const float* astro = (const float*)d_in[3];
  const float* Wq = (const float*)d_in[4];  const float* bq = (const float*)d_in[5];
  const float* Wk = (const float*)d_in[6];  const float* bk = (const float*)d_in[7];
  const float* Wv = (const float*)d_in[8];  const float* bv = (const float*)d_in[9];
  const float* Wo = (const float*)d_in[10]; const float* bo = (const float*)d_in[11];
  const float* role = (const float*)d_in[12];
  const float* Wr1 = (const float*)d_in[13]; const float* br1 = (const float*)d_in[14];
  const float* Wr2 = (const float*)d_in[15];
  const float* ascale = (const float*)d_in[16];
  float* out = (float*)d_out;
  float* new_state_out = out + (long)B_SZ * T_Q * D_M;

  // workspace carve-up
  char* base = (char*)d_ws; size_t off = 0;
  auto alloc = [&](size_t bytes) -> void* {
    void* p = base + off; off = (off + bytes + 255) & ~(size_t)255; return p;
  };
  bf16* Wq_b  = (bf16*)alloc((size_t)D_M * D_M * 2);
  bf16* Wk_b  = (bf16*)alloc((size_t)D_M * D_M * 2);
  bf16* Wv_b  = (bf16*)alloc((size_t)D_M * D_M * 2);
  bf16* Wo_b  = (bf16*)alloc((size_t)D_M * D_M * 2);
  bf16* Wr1_b = (bf16*)alloc((size_t)D_M * H_H * 2);
  bf16* Wr2_b = (bf16*)alloc((size_t)H_H * R_R * 2);
  bf16* role_b = (bf16*)alloc((size_t)R_R * D_M * 2);
  bf16* dft_f = (bf16*)alloc((size_t)D_M * 2 * D_M * 2);      // 1024 x 2048
  bf16* dft_i = (bf16*)alloc((size_t)2 * D_M * D_M * 2);      // 2048 x 1024
  float* Qf   = (float*)alloc((size_t)N_TOK * D_M * 4);
  bf16* Kb    = (bf16*)alloc((size_t)N_TOK * D_M * 2);
  bf16* Vb    = (bf16*)alloc((size_t)N_TOK * D_M * 2);
  bf16* qn    = (bf16*)alloc((size_t)N_TOK * D_M * 2);
  bf16* Hbuf  = (bf16*)alloc((size_t)N_TOK * H_H * 2);
  float* logits = (float*)alloc((size_t)N_TOK * R_R * 4);
  bf16* wts   = (bf16*)alloc((size_t)N_TOK * R_R * 2);
  bf16* rvec  = (bf16*)alloc((size_t)N_TOK * D_M * 2);
  bf16* KF    = (bf16*)alloc((size_t)N_TOK * 2 * D_M * 2);
  bf16* RF    = (bf16*)alloc((size_t)N_TOK * 2 * D_M * 2);
  bf16* Pb    = (bf16*)alloc((size_t)N_TOK * 2 * D_M * 2);
  float* Kbound = (float*)alloc((size_t)N_TOK * D_M * 4);
  bf16* kn    = (bf16*)alloc((size_t)N_TOK * D_M * 2);
  bf16* scores = (bf16*)alloc((size_t)B_SZ * T_Q * T_K * 2);
  float* rs   = (float*)alloc((size_t)N_TOK * 4);
  bf16* ctx   = (bf16*)alloc((size_t)N_TOK * D_M * 2);
  float* ssum = (float*)alloc(256);
  float* tau  = (float*)alloc(256);

  dim3 blk(256);
  auto gg = [](int M, int N, int bat) { return dim3(N / 64, M / 128, bat); };

  // weight conversions + constants
  k_convert<<<(D_M * D_M / 4 + 255) / 256, 256, 0, stream>>>(Wq, Wq_b, (long)D_M * D_M);
  k_convert<<<(D_M * D_M / 4 + 255) / 256, 256, 0, stream>>>(Wk, Wk_b, (long)D_M * D_M);
  k_convert<<<(D_M * D_M / 4 + 255) / 256, 256, 0, stream>>>(Wv, Wv_b, (long)D_M * D_M);
  k_convert<<<(D_M * D_M / 4 + 255) / 256, 256, 0, stream>>>(Wo, Wo_b, (long)D_M * D_M);
  k_convert<<<(D_M * H_H / 4 + 255) / 256, 256, 0, stream>>>(Wr1, Wr1_b, (long)D_M * H_H);
  k_convert<<<(H_H * R_R / 4 + 255) / 256, 256, 0, stream>>>(Wr2, Wr2_b, (long)H_H * R_R);
  k_role_norm<<<R_R, 256, 0, stream>>>(role, role_b);
  k_gen_dft<<<3072, 256, 0, stream>>>(dft_f, dft_i);
  k_zero<<<1, 256, 0, stream>>>(ssum, 8);

  // Q/K/V projections
  gemm_wmma<false, false, false, EPI_BIAS><<<gg(N_TOK, D_M, 1), blk, 0, stream>>>(
      q_in, Wq_b, Qf, N_TOK, D_M, D_M, 0, 0, 0, bq, nullptr, nullptr, 0);
  gemm_wmma<false, false, true, EPI_BIAS><<<gg(N_TOK, D_M, 1), blk, 0, stream>>>(
      k_in, Wk_b, Kb, N_TOK, D_M, D_M, 0, 0, 0, bk, nullptr, nullptr, 0);
  gemm_wmma<false, false, true, EPI_BIAS><<<gg(N_TOK, D_M, 1), blk, 0, stream>>>(
      v_in, Wv_b, Vb, N_TOK, D_M, D_M, 0, 0, 0, bv, nullptr, nullptr, 0);

  // q_n + surprise -> astro state -> tau
  k_rownorm<<<N_TOK, 256, 0, stream>>>(Qf, qn, ssum, T_Q);
  k_state<<<1, 32, 0, stream>>>(ssum, astro, ascale, new_state_out, tau);

  // role MLP + softmax + role vectors
  gemm_wmma<true, false, true, EPI_BIAS_RELU><<<gg(N_TOK, H_H, 1), blk, 0, stream>>>(
      Kb, Wr1_b, Hbuf, N_TOK, H_H, D_M, 0, 0, 0, br1, nullptr, nullptr, 0);
  gemm_wmma<true, false, false, EPI_NONE><<<gg(N_TOK, R_R, 1), blk, 0, stream>>>(
      Hbuf, Wr2_b, logits, N_TOK, R_R, H_H, 0, 0, 0, nullptr, nullptr, nullptr, 0);
  k_softmax64<<<N_TOK, 64, 0, stream>>>(logits, wts);
  gemm_wmma<true, false, true, EPI_NONE><<<gg(N_TOK, D_M, 1), blk, 0, stream>>>(
      wts, role_b, rvec, N_TOK, D_M, R_R, 0, 0, 0, nullptr, nullptr, nullptr, 0);

  // holographic binding via DFT-as-GEMM
  gemm_wmma<true, false, true, EPI_NONE><<<gg(N_TOK, 2 * D_M, 1), blk, 0, stream>>>(
      Kb, dft_f, KF, N_TOK, 2 * D_M, D_M, 0, 0, 0, nullptr, nullptr, nullptr, 0);
  gemm_wmma<true, false, true, EPI_NONE><<<gg(N_TOK, 2 * D_M, 1), blk, 0, stream>>>(
      rvec, dft_f, RF, N_TOK, 2 * D_M, D_M, 0, 0, 0, nullptr, nullptr, nullptr, 0);
  k_cmul<<<(N_TOK * D_M) / 256, 256, 0, stream>>>(KF, RF, Pb);
  gemm_wmma<true, false, false, EPI_NONE><<<gg(N_TOK, D_M, 1), blk, 0, stream>>>(
      Pb, dft_i, Kbound, N_TOK, D_M, 2 * D_M, 0, 0, 0, nullptr, nullptr, nullptr, 0);
  k_rownorm<<<N_TOK, 256, 0, stream>>>(Kbound, kn, nullptr, T_Q);

  // attention: scores (Epanechnikov epilogue, kn rows = transposed-B layout, async path),
  // rowsum, attn@V
  gemm_wmma<true, true, true, EPI_SCORE><<<gg(T_Q, T_K, B_SZ), blk, 0, stream>>>(
      qn, kn, scores, T_Q, T_K, D_M,
      (long)T_Q * D_M, (long)T_K * D_M, (long)T_Q * T_K, nullptr, tau, nullptr, 0);
  k_rowscale<<<N_TOK, 256, 0, stream>>>(scores, rs);
  gemm_wmma<true, false, true, EPI_ROWSCALE><<<gg(T_Q, D_M, B_SZ), blk, 0, stream>>>(
      scores, Vb, ctx, T_Q, D_M, T_K,
      (long)T_Q * T_K, (long)T_K * D_M, (long)T_Q * D_M, nullptr, nullptr, rs, T_Q);

  // output projection -> d_out (f32)
  gemm_wmma<true, false, false, EPI_BIAS><<<gg(N_TOK, D_M, 1), blk, 0, stream>>>(
      ctx, Wo_b, out, N_TOK, D_M, D_M, 0, 0, 0, bo, nullptr, nullptr, 0);
}
